// ImprovedCrossScaleGNN_85117661872507
// MI455X (gfx1250) — matrix operand order repaired and checked by reading, not compiled, not run on hardware
//
#include <hip/hip_runtime.h>
#include <cstdint>
#include <cstddef>

// ---------------- problem constants ----------------
#define NNODES 50000
#define NEDGES 600000
#define NCOMM  500
#define CEDGES 8000
#define NHID   128
#define NCLASS 10

// ---------------- CDNA5 WMMA types ----------------
typedef __attribute__((ext_vector_type(16))) __bf16 v16bf;
typedef __attribute__((ext_vector_type(8)))  float  v8f;

union FragAB { v16bf v; uint4 q[2]; };

static __device__ __forceinline__ __bf16 f2bf(float f) {
    unsigned u = __builtin_bit_cast(unsigned, f);
    unsigned r = (u + 0x7FFFu + ((u >> 16) & 1u)) >> 16;   // round-to-nearest-even
    unsigned short s = (unsigned short)r;
    return __builtin_bit_cast(__bf16, s);
}

// =====================================================================
// Generic bf16 WMMA GEMM:  C[M,Ncols] = A[M,128] * Wt[Ncols,128]^T
//   (Wt stored row-major as [ncol][k]  ==  B[k][ncol])
//   optional bias[Ncols], optional residual[M,Ncols]; f32 accumulate.
// Block: (32, WY) waves; wave -> one 16x16 tile (col tile = by*WY + wy).
// A tile (16x128 bf16) staged once per block in LDS via the CDNA5
// async global->LDS DMA path (ASYNCcnt), then 4x v_wmma_f32_16x16x32_bf16.
// =====================================================================
__global__ void __launch_bounds__(256) k_gemm_bf16(
    const __bf16* __restrict__ A, const __bf16* __restrict__ Wt,
    float* __restrict__ C, const float* __restrict__ bias,
    const float* __restrict__ resid, int M, int Ncols)
{
    __shared__ alignas(16) __bf16 ldsA[16 * 128];
    const int lane = threadIdx.x;                 // 0..31
    const int wy   = threadIdx.y;
    const int nthr = blockDim.x * blockDim.y;
    const int tid  = wy * 32 + lane;
    const int rowBase = blockIdx.x * 16;

    // stage A tile: 16 rows x 16 chunks of 16B, async DMA straight into LDS
    for (int idx = tid; idx < 256; idx += nthr) {
        const int r = idx >> 4, c = idx & 15;
        const int row = rowBase + r;
        __bf16* lp = &ldsA[r * 128 + c * 8];
        if (row < M) {
            const __bf16* gp = A + (size_t)row * 128 + c * 8;
            const unsigned lds_off = (unsigned)(size_t)lp;  // low 32 bits = LDS byte offset
            asm volatile("global_load_async_to_lds_b128 %0, %1, off"
                         :: "v"(lds_off), "v"(gp) : "memory");
        } else {
            *(uint4*)lp = make_uint4(0u, 0u, 0u, 0u);       // pad rows (comm GEMM tail)
        }
    }
    asm volatile("s_wait_asynccnt 0x0" ::: "memory");
    __syncthreads();

    const int ct   = blockIdx.y * blockDim.y + wy;   // column tile
    const int ncol = lane & 15;
    const int half = lane >> 4;
    const int j    = ct * 16 + ncol;                 // global output column

    v8f acc = {0.f, 0.f, 0.f, 0.f, 0.f, 0.f, 0.f, 0.f};
    const __bf16* arow = &ldsA[(lane & 15) * 128];   // A row m = lane&15
    const __bf16* brow = Wt + (size_t)j * 128;       // B column j (contiguous K)

    #pragma unroll
    for (int kc = 0; kc < 4; ++kc) {                 // K = 4 x 32
        FragAB a, b;
        // A 16x32 bf16 layout: v0..7 -> K = half*16 + 0..15 (contiguous)
        a.q[0] = *(const uint4*)(arow + kc * 32 + half * 16);
        a.q[1] = *(const uint4*)(arow + kc * 32 + half * 16 + 8);
        // B 32x16 bf16 layout: lane n = lane&15, K = half*16 + 0..15 (contiguous in Wt row)
        const uint4* pb = (const uint4*)(brow + kc * 32 + half * 16);
        b.q[0] = pb[0];
        b.q[1] = pb[1];
        acc = __builtin_amdgcn_wmma_f32_16x16x32_bf16(false, a.v, false, b.v,
                                                      (short)0, acc, false, false);
    }

    // C/D layout: element r -> row (half*8 + r), column j
    const int mBase = rowBase + half * 8;
    const float bj = bias ? bias[j] : 0.0f;
    if (rowBase + 16 <= M) {                          // full tile: unguarded stores
        if (resid) {
            #pragma unroll
            for (int r = 0; r < 8; ++r) {
                const size_t idx = (size_t)(mBase + r) * Ncols + j;
                C[idx] = acc[r] + bj + resid[idx];
            }
        } else {
            #pragma unroll
            for (int r = 0; r < 8; ++r) {
                const size_t idx = (size_t)(mBase + r) * Ncols + j;
                C[idx] = acc[r] + bj;
            }
        }
    } else {                                          // partial tile (M=500 comm tail)
        #pragma unroll
        for (int r = 0; r < 8; ++r) {
            const int m = mBase + r;
            if (m < M) {
                const size_t idx = (size_t)m * Ncols + j;
                float v = acc[r] + bj;
                if (resid) v += resid[idx];
                C[idx] = v;
            }
        }
    }
}

// ---------------- elementwise / prep kernels ----------------
__global__ void k_f32_to_bf16(const float* __restrict__ in, __bf16* __restrict__ out, int n) {
    int t = blockIdx.x * blockDim.x + threadIdx.x;
    if (t < n) out[t] = f2bf(in[t]);
}

// W[k][c] (row-major KxNc, math layout h@W) -> Wt[c][k] bf16
__global__ void k_transpose_bf16(const float* __restrict__ W, __bf16* __restrict__ Wt,
                                 int K, int Nc) {
    int t = blockIdx.x * blockDim.x + threadIdx.x;
    if (t >= K * Nc) return;
    int k = t / Nc, c = t % Nc;
    Wt[(size_t)c * K + k] = f2bf(W[t]);
}

__global__ void k_fill(float* __restrict__ p, float v, int n) {
    int t = blockIdx.x * blockDim.x + threadIdx.x;
    if (t < n) p[t] = v;
}

__global__ void k_deg_count(const int* __restrict__ dst, float* __restrict__ deg, int E) {
    int e = blockIdx.x * blockDim.x + threadIdx.x;
    if (e < E) atomicAdd(&deg[dst[e]], 1.0f);
}

__global__ void k_rsqrt(const float* __restrict__ deg, float* __restrict__ dinv, int n) {
    int t = blockIdx.x * blockDim.x + threadIdx.x;
    if (t < n) dinv[t] = rsqrtf(deg[t]);   // deg >= 1 (self-loop)
}

// agg = m * dinv^2 (self-loop) + bias
__global__ void k_agg_init(const float* __restrict__ m, const float* __restrict__ dinv,
                           const float* __restrict__ bias, float* __restrict__ agg, int n) {
    int t = blockIdx.x * blockDim.x + threadIdx.x;
    if (t >= n * 128) return;
    int r = t >> 7, f = t & 127;
    float di = dinv[r];
    agg[t] = m[t] * di * di + bias[f];
}

// 32 lanes per edge, float4 per lane: agg[dst] += m[src] * dinv[src]*dinv[dst]
__global__ void k_edge_scatter(const float* __restrict__ m, const int* __restrict__ src,
                               const int* __restrict__ dst, const float* __restrict__ dinv,
                               float* __restrict__ agg, int E) {
    int t = blockIdx.x * blockDim.x + threadIdx.x;
    int e = t >> 5;
    if (e >= E) return;
    int l = t & 31;
    int s = src[e], d = dst[e];
    float w = dinv[s] * dinv[d];
    const float4 mv = *(const float4*)(m + (size_t)s * 128 + l * 4);
    float* ad = agg + (size_t)d * 128 + l * 4;
    atomicAdd(ad + 0, mv.x * w);
    atomicAdd(ad + 1, mv.y * w);
    atomicAdd(ad + 2, mv.z * w);
    atomicAdd(ad + 3, mv.w * w);
}

// per-feature partial sums for BatchNorm (T must be multiple of NF)
__global__ void k_bn_partial(const float* __restrict__ X, float* __restrict__ sum,
                             float* __restrict__ sumsq, int n, int NF) {
    int t = blockIdx.x * blockDim.x + threadIdx.x;
    int T = gridDim.x * blockDim.x;
    int f = t % NF;
    int rstep = T / NF;
    float s = 0.f, ss = 0.f;
    for (int r = t / NF; r < n; r += rstep) {
        float v = X[(size_t)r * NF + f];
        s += v; ss += v * v;
    }
    atomicAdd(&sum[f], s);
    atomicAdd(&sumsq[f], ss);
}

__global__ void k_bn_finalize(const float* __restrict__ sum, const float* __restrict__ sumsq,
                              const float* __restrict__ g, const float* __restrict__ beta,
                              float* __restrict__ scale, float* __restrict__ shift,
                              int n, int NF) {
    int f = threadIdx.x;
    if (f >= NF) return;
    float inv_n = 1.0f / (float)n;
    float mean = sum[f] * inv_n;
    float var  = sumsq[f] * inv_n - mean * mean;
    float sc   = g[f] * rsqrtf(var + 1e-5f);
    scale[f] = sc;
    shift[f] = beta[f] - mean * sc;
}

// h = relu(agg*scale+shift) (+prev) ; optional bf16 mirror
__global__ void k_bn_apply(const float* __restrict__ agg, const float* __restrict__ scale,
                           const float* __restrict__ shift, const float* __restrict__ prev,
                           float* __restrict__ hout, __bf16* __restrict__ hbf,
                           int n, int NF) {
    int t = blockIdx.x * blockDim.x + threadIdx.x;
    if (t >= n * NF) return;
    int f = t % NF;
    float v = fmaxf(agg[t] * scale[f] + shift[f], 0.0f);
    if (prev) v += prev[t];
    hout[t] = v;
    if (hbf) hbf[t] = f2bf(v);
}

// X2[2n] = h_node[n], X2[2n+1] = h_comm[map[n]]   (bf16)
__global__ void k_build_x2(const float* __restrict__ hn, const float* __restrict__ hc,
                           const int* __restrict__ map, __bf16* __restrict__ X2, int n) {
    int t = blockIdx.x * blockDim.x + threadIdx.x;
    if (t >= n * 128) return;
    int r = t >> 7, f = t & 127;
    int cm = map[r];
    cm = cm < 0 ? 0 : (cm > NCOMM - 1 ? NCOMM - 1 : cm);
    X2[(size_t)(2 * r) * 128 + f]     = f2bf(hn[t]);
    X2[(size_t)(2 * r + 1) * 128 + f] = f2bf(hc[(size_t)cm * 128 + f]);
}

// fused 2-token, 8-head attention; emits mean-over-s of att@v directly (bf16)
__global__ void k_attention(const float* __restrict__ qkv, __bf16* __restrict__ omean, int n) {
    int t = blockIdx.x * blockDim.x + threadIdx.x;
    if (t >= n * 8) return;
    int nidx = t >> 3, h = t & 7;
    const float* r0 = qkv + (size_t)(2 * nidx) * 384 + h * 16;  // [q|k|v] row, s=0
    const float* r1 = r0 + 384;                                 // s=1
    float s00 = 0.f, s01 = 0.f, s10 = 0.f, s11 = 0.f;
    #pragma unroll
    for (int d = 0; d < 16; ++d) {
        float q0 = r0[d],       q1 = r1[d];
        float k0 = r0[128 + d], k1 = r1[128 + d];
        s00 += q0 * k0; s01 += q0 * k1; s10 += q1 * k0; s11 += q1 * k1;
    }
    const float sc = 0.25f;  // 1/sqrt(16)
    s00 *= sc; s01 *= sc; s10 *= sc; s11 *= sc;
    float m0 = fmaxf(s00, s01), m1 = fmaxf(s10, s11);
    float e00 = expf(s00 - m0), e01 = expf(s01 - m0);
    float e10 = expf(s10 - m1), e11 = expf(s11 - m1);
    float i0 = 1.0f / (e00 + e01), i1 = 1.0f / (e10 + e11);
    float c0 = 0.5f * (e00 * i0 + e10 * i1);   // mean over q of weight on v0
    float c1 = 0.5f * (e01 * i0 + e11 * i1);   // mean over q of weight on v1
    __bf16* out = omean + (size_t)nidx * 128 + h * 16;
    #pragma unroll
    for (int d = 0; d < 16; ++d)
        out[d] = f2bf(c0 * r0[256 + d] + c1 * r1[256 + d]);
}

// wave-per-row LayerNorm over 128, emits bf16
__global__ void __launch_bounds__(256) k_layernorm(const float* __restrict__ x,
                                                   const float* __restrict__ g,
                                                   const float* __restrict__ b,
                                                   __bf16* __restrict__ out, int n) {
    int wid = threadIdx.x >> 5, lane = threadIdx.x & 31;
    int row = blockIdx.x * 8 + wid;
    if (row >= n) return;
    const float* xr = x + (size_t)row * 128;
    float v[4]; float s = 0.f, ss = 0.f;
    #pragma unroll
    for (int i = 0; i < 4; ++i) { v[i] = xr[lane + 32 * i]; s += v[i]; ss += v[i] * v[i]; }
    #pragma unroll
    for (int o = 16; o >= 1; o >>= 1) { s += __shfl_xor(s, o, 32); ss += __shfl_xor(ss, o, 32); }
    float mean = s * (1.f / 128.f);
    float var  = ss * (1.f / 128.f) - mean * mean;
    float inv  = rsqrtf(var + 1e-5f);
    __bf16* orow = out + (size_t)row * 128;
    #pragma unroll
    for (int i = 0; i < 4; ++i) {
        int f = lane + 32 * i;
        orow[f] = f2bf((v[i] - mean) * inv * g[f] + b[f]);
    }
}

// BN+ReLU on z[64] -> logits[10] via W2 -> log_softmax
__global__ void k_classifier(const float* __restrict__ z, const float* __restrict__ scale,
                             const float* __restrict__ shift, const float* __restrict__ W2,
                             const float* __restrict__ b2, float* __restrict__ out, int n) {
    int r = blockIdx.x * blockDim.x + threadIdx.x;
    if (r >= n) return;
    float logit[NCLASS];
    #pragma unroll
    for (int j = 0; j < NCLASS; ++j) logit[j] = b2[j];
    const float* zr = z + (size_t)r * 64;
    for (int k = 0; k < 64; ++k) {
        float v = fmaxf(zr[k] * scale[k] + shift[k], 0.0f);
        #pragma unroll
        for (int j = 0; j < NCLASS; ++j) logit[j] += v * W2[k * NCLASS + j];
    }
    float m = logit[0];
    #pragma unroll
    for (int j = 1; j < NCLASS; ++j) m = fmaxf(m, logit[j]);
    float se = 0.f;
    #pragma unroll
    for (int j = 0; j < NCLASS; ++j) se += expf(logit[j] - m);
    float lse = logf(se) + m;
    float* orow = out + (size_t)r * NCLASS;
    #pragma unroll
    for (int j = 0; j < NCLASS; ++j) orow[j] = logit[j] - lse;
}

// ---------------- host ----------------
static inline int cdiv(long a, long b) { return (int)((a + b - 1) / b); }

extern "C" void kernel_launch(void* const* d_in, const int* in_sizes, int n_in,
                              void* d_out, int out_size, void* d_ws, size_t ws_size,
                              hipStream_t stream) {
    (void)in_sizes; (void)n_in; (void)out_size; (void)ws_size;

    // ---- inputs (setup_inputs order) ----
    const float* node_features = (const float*)d_in[0];
    const int*   nEdge         = (const int*)d_in[1];     // [2, NEDGES]
    const float* comm_features = (const float*)d_in[2];
    const int*   cEdge         = (const int*)d_in[3];     // [2, CEDGES]
    const int*   n2c           = (const int*)d_in[4];
    const float* node_W  = (const float*)d_in[5];
    const float* node_b  = (const float*)d_in[6];
    const float* node_g  = (const float*)d_in[7];
    const float* node_bb = (const float*)d_in[8];
    const float* comm_W  = (const float*)d_in[9];
    const float* comm_b  = (const float*)d_in[10];
    const float* comm_g  = (const float*)d_in[11];
    const float* comm_bb = (const float*)d_in[12];
    const float* attn_in_w  = (const float*)d_in[13];     // [384,128] (already [n][k])
    const float* attn_in_b  = (const float*)d_in[14];
    const float* attn_out_w = (const float*)d_in[15];     // [128,128] (already [n][k])
    const float* attn_out_b = (const float*)d_in[16];
    const float* ln_g   = (const float*)d_in[17];
    const float* ln_b   = (const float*)d_in[18];
    const float* cls_W1 = (const float*)d_in[19];         // [128,64] math layout
    const float* cls_b1 = (const float*)d_in[20];
    const float* cls_g  = (const float*)d_in[21];
    const float* cls_bb = (const float*)d_in[22];
    const float* cls_W2 = (const float*)d_in[23];         // [64,10]
    const float* cls_b2 = (const float*)d_in[24];
    float* out = (float*)d_out;

    const int* nSrc = nEdge;               const int* nDst = nEdge + NEDGES;
    const int* cSrc = cEdge;               const int* cDst = cEdge + CEDGES;

    // ---- workspace carve (256B aligned) ----
    char* w = (char*)d_ws;
    size_t off = 0;
    auto alloc = [&](size_t bytes) -> void* {
        void* p = w + off;
        off = (off + bytes + 255) & ~(size_t)255;
        return p;
    };
    float* NB0  = (float*)alloc((size_t)NNODES * 128 * 4);   // gemm out / scratch
    float* NB1  = (float*)alloc((size_t)NNODES * 128 * 4);   // h1 / h_node final
    float* NB2  = (float*)alloc((size_t)NNODES * 128 * 4);   // h2 / x
    float* aggN = (float*)alloc((size_t)NNODES * 128 * 4);   // agg / z
    float* QKV  = (float*)alloc((size_t)2 * NNODES * 384 * 4);
    float* CB0  = (float*)alloc((size_t)NCOMM * 128 * 4);
    float* CB1  = (float*)alloc((size_t)NCOMM * 128 * 4);
    float* CB2  = (float*)alloc((size_t)NCOMM * 128 * 4);
    float* aggC = (float*)alloc((size_t)NCOMM * 128 * 4);
    float* degN = (float*)alloc((size_t)NNODES * 4);
    float* dinvN= (float*)alloc((size_t)NNODES * 4);
    float* degC = (float*)alloc((size_t)NCOMM * 4);
    float* dinvC= (float*)alloc((size_t)NCOMM * 4);
    float* sumB = (float*)alloc(128 * 4);
    float* sumS = (float*)alloc(128 * 4);
    float* scB  = (float*)alloc(128 * 4);
    float* shB  = (float*)alloc(128 * 4);
    __bf16* hNbf  = (__bf16*)alloc((size_t)NNODES * 128 * 2);   // gcn input / o_mean
    __bf16* hFbf  = (__bf16*)alloc((size_t)NNODES * 128 * 2);   // LN output
    __bf16* hCbf  = (__bf16*)alloc((size_t)NCOMM * 128 * 2);
    __bf16* X2    = (__bf16*)alloc((size_t)2 * NNODES * 128 * 2);
    __bf16* nWt   = (__bf16*)alloc((size_t)3 * 128 * 128 * 2);
    __bf16* cWt   = (__bf16*)alloc((size_t)3 * 128 * 128 * 2);
    __bf16* inWb  = (__bf16*)alloc((size_t)384 * 128 * 2);
    __bf16* outWb = (__bf16*)alloc((size_t)128 * 128 * 2);
    __bf16* W1t   = (__bf16*)alloc((size_t)64 * 128 * 2);
    float* zBuf   = aggN;     // [NNODES,64] reuses agg region (dead by then)
    float* xBuf   = NB2;      // LN input reuses NB2 (h2 dead by then)
    __bf16* omean = hNbf;     // o_mean reuses gcn-input bf16 (dead by then)

    auto gemm = [&](const __bf16* A, const __bf16* Wt, float* C, const float* bias,
                    const float* resid, int M, int Ncols) {
        int tiles = Ncols / 16;
        int wy = tiles >= 8 ? 8 : tiles;
        dim3 blk(32, wy);
        dim3 grd(cdiv(M, 16), tiles / wy);
        k_gemm_bf16<<<grd, blk, 0, stream>>>(A, Wt, C, bias, resid, M, Ncols);
    };

    // ---- weight prep ----
    for (int i = 0; i < 3; ++i) {
        k_transpose_bf16<<<cdiv(128 * 128, 256), 256, 0, stream>>>(node_W + (size_t)i * 128 * 128, nWt + (size_t)i * 128 * 128, 128, 128);
        k_transpose_bf16<<<cdiv(128 * 128, 256), 256, 0, stream>>>(comm_W + (size_t)i * 128 * 128, cWt + (size_t)i * 128 * 128, 128, 128);
    }
    k_f32_to_bf16<<<cdiv(384 * 128, 256), 256, 0, stream>>>(attn_in_w, inWb, 384 * 128);
    k_f32_to_bf16<<<cdiv(128 * 128, 256), 256, 0, stream>>>(attn_out_w, outWb, 128 * 128);
    k_transpose_bf16<<<cdiv(128 * 64, 256), 256, 0, stream>>>(cls_W1, W1t, 128, 64);

    // ---- degrees / symmetric norm ----
    k_fill<<<cdiv(NNODES, 256), 256, 0, stream>>>(degN, 1.0f, NNODES);
    k_deg_count<<<cdiv(NEDGES, 256), 256, 0, stream>>>(nDst, degN, NEDGES);
    k_rsqrt<<<cdiv(NNODES, 256), 256, 0, stream>>>(degN, dinvN, NNODES);
    k_fill<<<cdiv(NCOMM, 256), 256, 0, stream>>>(degC, 1.0f, NCOMM);
    k_deg_count<<<cdiv(CEDGES, 256), 256, 0, stream>>>(cDst, degC, CEDGES);
    k_rsqrt<<<cdiv(NCOMM, 256), 256, 0, stream>>>(degC, dinvC, NCOMM);

    // ---- one GCN layer ----
    auto gcn_layer = [&](int n, int E, const __bf16* hbf_in, const __bf16* Wt,
                         const float* bias, const float* bng, const float* bnb,
                         const int* esrc, const int* edst, const float* dinv,
                         float* mbuf, float* agg, const float* prev,
                         float* hout, __bf16* hbf_out) {
        gemm(hbf_in, Wt, mbuf, nullptr, nullptr, n, 128);
        int tot = n * 128;
        k_agg_init<<<cdiv(tot, 256), 256, 0, stream>>>(mbuf, dinv, bias, agg, n);
        k_edge_scatter<<<cdiv((long)E * 32, 256), 256, 0, stream>>>(mbuf, esrc, edst, dinv, agg, E);
        hipMemsetAsync(sumB, 0, 128 * 4, stream);
        hipMemsetAsync(sumS, 0, 128 * 4, stream);
        k_bn_partial<<<128, 256, 0, stream>>>(agg, sumB, sumS, n, 128);
        k_bn_finalize<<<1, 128, 0, stream>>>(sumB, sumS, bng, bnb, scB, shB, n, 128);
        k_bn_apply<<<cdiv(tot, 256), 256, 0, stream>>>(agg, scB, shB, prev, hout, hbf_out, n, 128);
    };

    // ---- node GCN stack ----
    k_f32_to_bf16<<<cdiv(NNODES * 128, 256), 256, 0, stream>>>(node_features, hNbf, NNODES * 128);
    gcn_layer(NNODES, NEDGES, hNbf, nWt,              node_b,        node_g,        node_bb,        nSrc, nDst, dinvN, NB0, aggN, nullptr, NB1, hNbf);
    gcn_layer(NNODES, NEDGES, hNbf, nWt + 128 * 128,  node_b + 128,  node_g + 128,  node_bb + 128,  nSrc, nDst, dinvN, NB0, aggN, NB1,     NB2, hNbf);
    gcn_layer(NNODES, NEDGES, hNbf, nWt + 2*128*128,  node_b + 256,  node_g + 256,  node_bb + 256,  nSrc, nDst, dinvN, NB0, aggN, NB2,     NB1, nullptr);
    // h_node final -> NB1

    // ---- community GCN stack ----
    k_f32_to_bf16<<<cdiv(NCOMM * 128, 256), 256, 0, stream>>>(comm_features, hCbf, NCOMM * 128);
    gcn_layer(NCOMM, CEDGES, hCbf, cWt,              comm_b,        comm_g,        comm_bb,        cSrc, cDst, dinvC, CB0, aggC, nullptr, CB1, hCbf);
    gcn_layer(NCOMM, CEDGES, hCbf, cWt + 128 * 128,  comm_b + 128,  comm_g + 128,  comm_bb + 128,  cSrc, cDst, dinvC, CB0, aggC, CB1,     CB2, hCbf);
    gcn_layer(NCOMM, CEDGES, hCbf, cWt + 2*128*128,  comm_b + 256,  comm_g + 256,  comm_bb + 256,  cSrc, cDst, dinvC, CB0, aggC, CB2,     CB1, nullptr);
    // h_comm final -> CB1

    // ---- cross-scale attention ----
    k_build_x2<<<cdiv(NNODES * 128, 256), 256, 0, stream>>>(NB1, CB1, n2c, X2, NNODES);
    gemm(X2, inWb, QKV, attn_in_b, nullptr, 2 * NNODES, 384);           // qkv
    k_attention<<<cdiv(NNODES * 8, 256), 256, 0, stream>>>(QKV, omean, NNODES);
    gemm(omean, outWb, xBuf, attn_out_b, NB1, NNODES, 128);             // out-proj + bo + h_node

    // ---- gate LayerNorm ----
    k_layernorm<<<cdiv(NNODES, 8), 256, 0, stream>>>(xBuf, ln_g, ln_b, hFbf, NNODES);

    // ---- classifier ----
    gemm(hFbf, W1t, zBuf, cls_b1, nullptr, NNODES, 64);
    hipMemsetAsync(sumB, 0, 128 * 4, stream);
    hipMemsetAsync(sumS, 0, 128 * 4, stream);
    k_bn_partial<<<128, 256, 0, stream>>>(zBuf, sumB, sumS, NNODES, 64);
    k_bn_finalize<<<1, 64, 0, stream>>>(sumB, sumS, cls_g, cls_bb, scB, shB, NNODES, 64);
    k_classifier<<<cdiv(NNODES, 256), 256, 0, stream>>>(zBuf, scB, shB, cls_W2, cls_b2, out, NNODES);
}
